// SelfAttention_87729001988743
// MI455X (gfx1250) — compile-verified, matrix operand order checked by reading, not run on hardware
//
#include <hip/hip_runtime.h>

// ---------------- problem constants ----------------
#define EMBED 768
#define SEQ   2048
#define BATCH 8
#define NROW  (BATCH * SEQ)          // 16384 rows of x / Q / K
#define KSTEPS (EMBED / 32)          // 24 WMMA k-steps (K=32 per wmma)
#define SCALE 0.03608439182435161f   // 1/sqrt(768)

// ---------------- CDNA5 WMMA types ----------------
typedef __attribute__((ext_vector_type(16))) __bf16         v16bf;
typedef __attribute__((ext_vector_type(8)))  float          v8f;
typedef __attribute__((ext_vector_type(4)))  float          v4f;
typedef __attribute__((ext_vector_type(8)))  unsigned short v8us;

union FragBF {            // 16 bf16 elements == 8 VGPRs
    v16bf bf;
    v8us  h[2];           // h[0] = elements 0..7, h[1] = 8..15
};

// hardware fp32 -> bf16 (v_cvt_*bf16_f32), not integer emulation
static __device__ __forceinline__ unsigned short f2bf(float f) {
    union { __bf16 b; unsigned short u; } cv;
    cv.b = (__bf16)f;
    return cv.u;
}

static __device__ __forceinline__ v8us cvt8(const float* p) {
    float4 a = *(const float4*)p;
    float4 b = *(const float4*)(p + 4);
    v8us r;
    r[0]=f2bf(a.x); r[1]=f2bf(a.y); r[2]=f2bf(a.z); r[3]=f2bf(a.w);
    r[4]=f2bf(b.x); r[5]=f2bf(b.y); r[6]=f2bf(b.z); r[7]=f2bf(b.w);
    return r;
}

// ---------------- kernel 0: W[k][n] fp32 -> Wt[n][k] bf16 ----------------
__global__ __launch_bounds__(256)
void convert_w_kernel(const float* __restrict__ Wq, const float* __restrict__ Wk,
                      unsigned short* __restrict__ WqT, unsigned short* __restrict__ WkT) {
    int tid = blockIdx.x * 256 + threadIdx.x;
    const int N = EMBED * EMBED;
    if (tid < N) {
        int n = tid / EMBED, k = tid % EMBED;
        WqT[tid] = f2bf(Wq[k * EMBED + n]);
    } else if (tid < 2 * N) {
        int t = tid - N;
        int n = t / EMBED, k = t % EMBED;
        WkT[t] = f2bf(Wk[k * EMBED + n]);
    }
}

// ---------------- kernel 1: Q = bf16(x@Wq + bq), K = bf16(x@Wk + bk) ------
// 8 waves per block share one 16-row x tile staged (and bf16-converted) in LDS.
// Each wave: one 16x16 output tile, 24 x v_wmma_f32_16x16x32_bf16.
// Grid: 1024 m-tiles x 12 parts; wave handles nt_global = part*8 + wave (0..95):
//   nt 0..47 -> Q matrix, nt 48..95 -> K matrix.
__global__ __launch_bounds__(256)
void qk_proj_kernel(const float* __restrict__ x,
                    const unsigned short* __restrict__ WqT,
                    const unsigned short* __restrict__ WkT,
                    const float* __restrict__ bq, const float* __restrict__ bk,
                    unsigned short* __restrict__ Qout, unsigned short* __restrict__ Kout) {
    __shared__ __align__(16) unsigned short xtile[16 * EMBED];   // 24 KB

    const int mt   = blockIdx.x / 12;
    const int part = blockIdx.x % 12;
    const int t    = threadIdx.x;

    // ---- stage x tile: 16 x 768 fp32 -> bf16 in LDS (6 x 8 elems / thread) ----
    {
        const float* xb = x + (size_t)(mt * 16) * EMBED;
#pragma unroll
        for (int i = 0; i < 6; ++i) {
            int g   = t + i * 256;               // 0..1535 granules of 8 elems
            int row = g / 96, gi = g % 96;
            *(v8us*)(xtile + g * 8) = cvt8(xb + (size_t)row * EMBED + gi * 8);
        }
    }
    __syncthreads();

    const int lane = t & 31;
    const int wv   = t >> 5;
    const int lm   = lane & 15;       // row (A) / col (B,C)
    const int hl   = lane >> 4;       // k-half select

    int nt_global = part * 8 + wv;                 // 0..95
    int which = nt_global >= (EMBED / 16);         // 0 = Q, 1 = K
    int nt = which ? nt_global - (EMBED / 16) : nt_global;

    const unsigned short* Wt   = which ? WkT : WqT;
    const float*          bias = which ? bk  : bq;
    unsigned short*       out  = which ? Kout : Qout;

    const unsigned short* arow = xtile + lm * EMBED + hl * 8;           // LDS
    const unsigned short* wrow = Wt + (size_t)(nt * 16 + lm) * EMBED + hl * 8;

    v8f c = {0.f,0.f,0.f,0.f,0.f,0.f,0.f,0.f};
    for (int kk = 0; kk < KSTEPS; ++kk) {
        FragBF a, b;
        a.h[0] = *(const v8us*)(arow + kk * 32);
        a.h[1] = *(const v8us*)(arow + kk * 32 + 16);
        b.h[0] = *(const v8us*)(wrow + kk * 32);
        b.h[1] = *(const v8us*)(wrow + kk * 32 + 16);
        c = __builtin_amdgcn_wmma_f32_16x16x32_bf16(false, a.bf, false, b.bf,
                                                    (short)0, c, false, false);
    }

    float bval = bias[nt * 16 + lm];
    unsigned short* orow = out + (size_t)(mt * 16 + 8 * hl) * EMBED + nt * 16 + lm;
#pragma unroll
    for (int r = 0; r < 8; ++r)
        orow[(size_t)r * EMBED] = f2bf(c[r] + bval);
}

// ---------------- kernel 2: fused scores + softmax ----------------
// 1 workgroup per (batch, 16 query rows): 16x2048 score row kept in LDS.
#define QTILE_BYTES (16 * EMBED * 2)      // 24576
#define SROW 2052                          // fp32 dwords per score row (16B aligned, bank-friendly)
#define SMEM_BYTES (QTILE_BYTES + 16 * SROW * 4)   // 155904 B (< 320KB/WGP)

__global__ __launch_bounds__(256)
void attn_scores_kernel(const unsigned short* __restrict__ Q,
                        const unsigned short* __restrict__ K,
                        float* __restrict__ out) {
    extern __shared__ __align__(16) char smem[];
    const int b  = blockIdx.x >> 7;        // batch
    const int mt = blockIdx.x & 127;       // query 16-row tile
    const int q0 = mt * 16;
    const int t  = threadIdx.x;

    // ---- stage Q tile (16 x 768 bf16) into LDS: 6 x 16B per thread ----
    {
        const unsigned short* qb = Q + (size_t)(b * SEQ + q0) * EMBED;
#pragma unroll
        for (int i = 0; i < 6; ++i) {
            int g   = t + i * 256;              // 0..1535 granules of 16B
            int row = g / 96, gi = g % 96;
            uint4 v = *(const uint4*)(qb + (size_t)row * EMBED + gi * 8);
            *(uint4*)(smem + g * 16) = v;
        }
    }
    __syncthreads();

    const int lane = t & 31;
    const int wv   = t >> 5;
    const int lm   = lane & 15;
    const int hl   = lane >> 4;

    // each wave owns 256 contiguous key columns: [wv*256, wv*256+256)
    v8f acc[16];
    const v8f zero = {0.f,0.f,0.f,0.f,0.f,0.f,0.f,0.f};
#pragma unroll
    for (int j = 0; j < 16; ++j) acc[j] = zero;

    const unsigned short* kbase = K + (size_t)(b * SEQ + wv * 256 + lm) * EMBED + hl * 8;

    for (int kk = 0; kk < KSTEPS; ++kk) {
        FragBF a;   // A fragment from LDS (shared by 16 WMMAs)
        a.h[0] = *(const v8us*)(smem + lm * 1536 + kk * 64 + hl * 16);
        a.h[1] = *(const v8us*)(smem + lm * 1536 + kk * 64 + hl * 16 + 32);
#pragma unroll
        for (int j = 0; j < 16; ++j) {
            FragBF bb;
            const unsigned short* kp = kbase + (size_t)j * 16 * EMBED + kk * 32;
            bb.h[0] = *(const v8us*)kp;
            bb.h[1] = *(const v8us*)(kp + 16);
            acc[j] = __builtin_amdgcn_wmma_f32_16x16x32_bf16(false, a.bf, false, bb.bf,
                                                             (short)0, acc[j], false, false);
        }
    }

    // ---- spill scaled scores to LDS ----
    float* sc = (float*)(smem + QTILE_BYTES);
#pragma unroll
    for (int j = 0; j < 16; ++j) {
#pragma unroll
        for (int r = 0; r < 8; ++r)
            sc[(r + 8 * hl) * SROW + wv * 256 + j * 16 + lm] = acc[j][r] * SCALE;
    }
    __syncthreads();

    // ---- softmax: 16 threads per row, 4-dword interleave (coalesced) ----
    const int row = t >> 4;
    const int sub = t & 15;
    float* srow = sc + row * SROW;

    float m = -3.402823466e38f;
#pragma unroll 8
    for (int k = 0; k < 32; ++k) {
        v4f v = *(const v4f*)(srow + sub * 4 + k * 64);
        m = fmaxf(m, fmaxf(fmaxf(v.x, v.y), fmaxf(v.z, v.w)));
    }
    for (int d = 1; d < 16; d <<= 1) m = fmaxf(m, __shfl_xor(m, d, 32));

    float s = 0.f;
#pragma unroll 8
    for (int k = 0; k < 32; ++k) {
        v4f v = *(v4f*)(srow + sub * 4 + k * 64);
        v.x = __expf(v.x - m); v.y = __expf(v.y - m);
        v.z = __expf(v.z - m); v.w = __expf(v.w - m);
        s += v.x + v.y + v.z + v.w;
        *(v4f*)(srow + sub * 4 + k * 64) = v;
    }
    for (int d = 1; d < 16; d <<= 1) s += __shfl_xor(s, d, 32);
    float inv = 1.0f / s;

    // nontemporal final store: 128 MB streamed once, keep L2 for K
    float* orow = out + (size_t)(b * SEQ + q0 + row) * SEQ;
#pragma unroll 8
    for (int k = 0; k < 32; ++k) {
        v4f v = *(const v4f*)(srow + sub * 4 + k * 64);
        v *= inv;
        __builtin_nontemporal_store(v, (v4f*)(orow + sub * 4 + k * 64));
    }
}

// ---------------- host launcher ----------------
extern "C" void kernel_launch(void* const* d_in, const int* in_sizes, int n_in,
                              void* d_out, int out_size, void* d_ws, size_t ws_size,
                              hipStream_t stream) {
    const float* x  = (const float*)d_in[0];
    const float* Wq = (const float*)d_in[1];
    const float* bq = (const float*)d_in[2];
    const float* Wk = (const float*)d_in[3];
    const float* bk = (const float*)d_in[4];
    // d_in[5]/d_in[6] (Wv, bv) are dead in the reference — skipped entirely.
    float* out = (float*)d_out;

    // workspace layout (bf16): Q[16384*768] | K[16384*768] | WqT[768^2] | WkT[768^2]
    const size_t QK = (size_t)NROW * EMBED;            // elements
    unsigned short* Qws = (unsigned short*)d_ws;
    unsigned short* Kws = Qws + QK;
    unsigned short* WqT = Kws + QK;
    unsigned short* WkT = WqT + (size_t)EMBED * EMBED; // total ~50.3 MB

    // 0) weight transpose + bf16 convert
    convert_w_kernel<<<(2 * EMBED * EMBED + 255) / 256, 256, 0, stream>>>(Wq, Wk, WqT, WkT);

    // 1) Q/K projections: 1024 m-tiles x 12 parts, 8 wave-tiles per block
    qk_proj_kernel<<<(NROW / 16) * 12, 256, 0, stream>>>(
        x, WqT, WkT, bq, bk, Qws, Kws);

    // 2) fused scores + softmax (needs >64KB dynamic LDS — CDNA5 WGP has 320KB)
    (void)hipFuncSetAttribute((const void*)attn_scores_kernel,
                              hipFuncAttributeMaxDynamicSharedMemorySize, SMEM_BYTES);
    attn_scores_kernel<<<BATCH * (SEQ / 16), 256, SMEM_BYTES, stream>>>(Qws, Kws, out);
}
